// GATv2Heteroscedastic_54580444397939
// MI455X (gfx1250) — matrix operand order repaired and checked by reading, not compiled, not run on hardware
//
#include <hip/hip_runtime.h>
#include <hip/hip_bf16.h>

#define NNODE  20000
#define NEDGE  160000
#define ETOT   (NEDGE + NNODE)   // self loops appended
#define NBATCH 64
#define NFEAT  128
#define NHID   64
#define NHEADS 16
#define HDIM   (NHEADS * NHID)   // 1024
#define NCLASS 10

typedef __attribute__((ext_vector_type(16))) __bf16 v16bf;
typedef __attribute__((ext_vector_type(8)))  __bf16 v8bf;
typedef __attribute__((ext_vector_type(8)))  float  v8f;

__device__ __forceinline__ float lrelu(float v, float s) { return v >= 0.0f ? v : s * v; }

__device__ __forceinline__ void edge_sd(const int* __restrict__ ei, int e, int& s, int& d) {
  if (e < NEDGE) { s = ei[e]; d = ei[NEDGE + e]; }
  else           { s = d = e - NEDGE; }
}

// float atomic max via int/uint monotonic punning (valid with -inf init)
__device__ __forceinline__ void atomicMaxF(float* addr, float v) {
  if (v >= 0.0f) atomicMax((int*)addr, __float_as_int(v));
  else           atomicMin((unsigned int*)addr, __float_as_uint(v));
}

// ---------------- utility kernels ----------------

__global__ void k_fill(float* __restrict__ p, float v, int n) {
  int i = blockIdx.x * blockDim.x + threadIdx.x;
  if (i < n) p[i] = v;
}

__global__ void k_cast_bf16(const float* __restrict__ s, __bf16* __restrict__ d, int n) {
  int i = blockIdx.x * blockDim.x + threadIdx.x;
  if (i < n) d[i] = (__bf16)s[i];
}

// W [K, Ncols] f32 row-major  ->  Wt [Ncols, K] bf16 row-major
__global__ void k_transpose_cast(const float* __restrict__ W, __bf16* __restrict__ Wt,
                                 int K, int Ncols) {
  int i = blockIdx.x * blockDim.x + threadIdx.x;
  if (i < K * Ncols) {
    int k = i / Ncols, c = i % Ncols;
    Wt[(size_t)c * K + k] = (__bf16)W[i];
  }
}

// ---------------- WMMA GEMM: C[M,Nc] = A[M,K](bf16) * Bt[Nc,K]^T(bf16) + bias ----------------
// One wave (32 lanes) per 16x16 output tile; K-loop of v_wmma_f32_16x16x32_bf16.
__global__ void k_gemm_wmma(const __bf16* __restrict__ A, const __bf16* __restrict__ Bt,
                            const float* __restrict__ bias, float* __restrict__ C,
                            int M, int Nc, int K) {
  const int lane = threadIdx.x;                         // 0..31
  const int tile = blockIdx.x * blockDim.y + threadIdx.y;
  const int tilesN = Nc >> 4;
  const int total  = (M >> 4) * tilesN;
  if (tile >= total) return;                            // wave-uniform: EXEC stays all-1
  const int row = (tile / tilesN) << 4;
  const int col = (tile % tilesN) << 4;

  // A fragment (16x32 bf16): lanes 0-15 hold M=lane, K 0..15; lanes 16-31 K 16..31
  const __bf16* ap = A + (size_t)(row + (lane & 15)) * K + ((lane >> 4) << 4);
  // B fragment (32x16 bf16): lanes 0-15 hold N=lane, K 0..15; lanes 16-31 K 16..31
  const __bf16* bp = Bt + (size_t)(col + (lane & 15)) * K + ((lane >> 4) << 4);

  v8f acc = {};
  for (int k0 = 0; k0 < K; k0 += 32) {
    v8bf a_lo = *(const v8bf*)(ap + k0);
    v8bf a_hi = *(const v8bf*)(ap + k0 + 8);
    v8bf b_lo = *(const v8bf*)(bp + k0);
    v8bf b_hi = *(const v8bf*)(bp + k0 + 8);
    v16bf a, b;
#pragma unroll
    for (int i = 0; i < 8; ++i) {
      a[i] = a_lo[i]; a[i + 8] = a_hi[i];
      b[i] = b_lo[i]; b[i + 8] = b_hi[i];
    }
    acc = __builtin_amdgcn_wmma_f32_16x16x32_bf16(false, a, false, b, (short)0, acc,
                                                  false, false);
  }
  // C/D layout: VGPR r -> M = r (lanes 0-15) / 8+r (lanes 16-31), N = lane&15
  const int n  = col + (lane & 15);
  const int m0 = row + ((lane >> 4) << 3);
  const float bv = bias[n];
#pragma unroll
  for (int r = 0; r < 8; ++r)
    C[(size_t)(m0 + r) * Nc + n] = acc[r] + bv;
}

// ---------------- edge-phase kernels ----------------

// one wave per (edge, head): logit = att_h . lrelu(xl[src,h,:] + xr[dst,h,:])
__global__ void k_logit(const float* __restrict__ xl, const float* __restrict__ xr,
                        const float* __restrict__ att, const int* __restrict__ ei,
                        float* __restrict__ logits, float* __restrict__ maxb) {
  int w = blockIdx.x * blockDim.y + threadIdx.y;
  if (w >= ETOT * NHEADS) return;
  int e = w >> 4, hd = w & 15;
  int s, d; edge_sd(ei, e, s, d);
  int lane = threadIdx.x;
  const float* pl = xl + (size_t)s * HDIM + hd * NHID;
  const float* pr = xr + (size_t)d * HDIM + hd * NHID;
  const float* pa = att + hd * NHID;
  float v0 = lrelu(pl[lane]      + pr[lane],      0.2f);
  float v1 = lrelu(pl[lane + 32] + pr[lane + 32], 0.2f);
  float p = pa[lane] * v0 + pa[lane + 32] * v1;
#pragma unroll
  for (int off = 16; off; off >>= 1) p += __shfl_xor(p, off, 32);
  if (lane == 0) {
    logits[w] = p;
    atomicMaxF(&maxb[d * NHEADS + hd], p);
  }
}

// one thread per (edge, head): ex = exp(logit - max[dst]); accumulate denominator
__global__ void k_exp(float* __restrict__ logits, const float* __restrict__ maxb,
                      float* __restrict__ den, const int* __restrict__ ei) {
  int i = blockIdx.x * blockDim.x + threadIdx.x;
  if (i >= ETOT * NHEADS) return;
  int e = i >> 4, hd = i & 15;
  int s, d; edge_sd(ei, e, s, d);
  float ex = __expf(logits[i] - maxb[d * NHEADS + hd]);
  logits[i] = ex;                            // reuse buffer: now holds exp()
  atomicAdd(&den[d * NHEADS + hd], ex);
}

// one wave per (edge, head): accum[dst,h,:] += alpha * xl[src,h,:]
__global__ void k_scatter(const float* __restrict__ exb, const float* __restrict__ den,
                          const float* __restrict__ xl, const int* __restrict__ ei,
                          float* __restrict__ accum) {
  int w = blockIdx.x * blockDim.y + threadIdx.y;
  if (w >= ETOT * NHEADS) return;
  int e = w >> 4, hd = w & 15;
  int s, d; edge_sd(ei, e, s, d);
  int lane = threadIdx.x;
  float alpha = exb[w] / den[d * NHEADS + hd];
  const float* pl = xl + (size_t)s * HDIM + hd * NHID;
  float* po = accum + (size_t)d * HDIM + hd * NHID;
  atomicAdd(&po[lane],      alpha * pl[lane]);
  atomicAdd(&po[lane + 32], alpha * pl[lane + 32]);
}

// one thread per (node, channel): head-mean + bias + outer LeakyReLU(0.01)
__global__ void k_node(const float* __restrict__ accum, const float* __restrict__ bias,
                       float* __restrict__ hout) {
  int i = blockIdx.x * blockDim.x + threadIdx.x;
  if (i >= NNODE * NHID) return;
  int n = i >> 6, c = i & 63;
  float sum = 0.0f;
#pragma unroll
  for (int h = 0; h < NHEADS; ++h) sum += accum[(size_t)n * HDIM + h * NHID + c];
  hout[i] = lrelu(sum * (1.0f / NHEADS) + bias[c], 0.01f);
}

// ---------------- pooling + heads ----------------

__global__ void k_pool(const float* __restrict__ h, const int* __restrict__ batch,
                       float* __restrict__ pooled, float* __restrict__ cnt) {
  int i = blockIdx.x * blockDim.x + threadIdx.x;
  if (i >= NNODE * NHID) return;
  int n = i >> 6, c = i & 63;
  int b = batch[n];
  atomicAdd(&pooled[b * NHID + c], h[i]);
  if (c == 0) atomicAdd(&cnt[b], 1.0f);
}

__global__ void k_head(const float* __restrict__ pooled, const float* __restrict__ cnt,
                       const float* __restrict__ Wc, const float* __restrict__ bc,
                       const float* __restrict__ Wv, const float* __restrict__ bv,
                       float* __restrict__ out) {
  int b = blockIdx.x, t = threadIdx.x;
  float inv = 1.0f / fmaxf(cnt[b], 1.0f);
  if (t < NCLASS) {
    float s = 0.0f;
    for (int c = 0; c < NHID; ++c) s += pooled[b * NHID + c] * inv * Wc[c * NCLASS + t];
    out[b * NCLASS + t] = s + bc[t];                 // TEMPERATURE == 1
  } else if (t == NCLASS) {
    float s = 0.0f;
    for (int c = 0; c < NHID; ++c) s += pooled[b * NHID + c] * inv * Wv[c];
    out[NBATCH * NCLASS + b] = s + bv[0];
  }
}

// ---------------- host orchestration ----------------

static inline int cdiv(int a, int b) { return (a + b - 1) / b; }

static void run_gat_layer(const __bf16* inbf, int K,
                          const __bf16* Wtl, const float* bl,
                          const __bf16* Wtr, const float* br,
                          const float* att, const float* bias,
                          const int* ei,
                          float* xl, float* xr, float* accum,
                          float* lbuf, float* maxb, float* den,
                          float* hout, hipStream_t stream) {
  const int tiles  = (NNODE / 16) * (HDIM / 16);
  const dim3 gblk(32, 4), gedge(32, 8);
  k_gemm_wmma<<<cdiv(tiles, 4), gblk, 0, stream>>>(inbf, Wtl, bl, xl, NNODE, HDIM, K);
  k_gemm_wmma<<<cdiv(tiles, 4), gblk, 0, stream>>>(inbf, Wtr, br, xr, NNODE, HDIM, K);

  k_fill<<<cdiv(NNODE * NHEADS, 256), 256, 0, stream>>>(maxb, -INFINITY, NNODE * NHEADS);
  k_fill<<<cdiv(NNODE * NHEADS, 256), 256, 0, stream>>>(den, 0.0f, NNODE * NHEADS);
  k_fill<<<cdiv(NNODE * HDIM, 256), 256, 0, stream>>>(accum, 0.0f, NNODE * HDIM);

  const int waves = ETOT * NHEADS;
  k_logit<<<cdiv(waves, 8), gedge, 0, stream>>>(xl, xr, att, ei, lbuf, maxb);
  k_exp<<<cdiv(waves, 256), 256, 0, stream>>>(lbuf, maxb, den, ei);
  k_scatter<<<cdiv(waves, 8), gedge, 0, stream>>>(lbuf, den, xl, ei, accum);
  k_node<<<cdiv(NNODE * NHID, 256), 256, 0, stream>>>(accum, bias, hout);
}

extern "C" void kernel_launch(void* const* d_in, const int* in_sizes, int n_in,
                              void* d_out, int out_size, void* d_ws, size_t ws_size,
                              hipStream_t stream) {
  (void)in_sizes; (void)n_in; (void)out_size; (void)ws_size;
  const float* x     = (const float*)d_in[0];
  const int*   ei    = (const int*)d_in[1];
  const int*   batch = (const int*)d_in[2];
  const float* W1l = (const float*)d_in[3],  *b1l  = (const float*)d_in[4];
  const float* W1r = (const float*)d_in[5],  *b1r  = (const float*)d_in[6];
  const float* att1= (const float*)d_in[7],  *bias1= (const float*)d_in[8];
  const float* W2l = (const float*)d_in[9],  *b2l  = (const float*)d_in[10];
  const float* W2r = (const float*)d_in[11], *b2r  = (const float*)d_in[12];
  const float* att2= (const float*)d_in[13], *bias2= (const float*)d_in[14];
  const float* Wc  = (const float*)d_in[15], *bc   = (const float*)d_in[16];
  const float* Wv  = (const float*)d_in[17], *bv   = (const float*)d_in[18];
  float* out = (float*)d_out;

  char* p = (char*)d_ws;
  auto walloc = [&](size_t bytes) -> void* {
    void* r = (void*)p;
    p += (bytes + 255) & ~(size_t)255;
    return r;
  };
  __bf16* xbf  = (__bf16*)walloc((size_t)NNODE * NFEAT * 2);
  __bf16* hbf  = (__bf16*)walloc((size_t)NNODE * NHID * 2);
  __bf16* Wt1l = (__bf16*)walloc((size_t)HDIM * NFEAT * 2);
  __bf16* Wt1r = (__bf16*)walloc((size_t)HDIM * NFEAT * 2);
  __bf16* Wt2l = (__bf16*)walloc((size_t)HDIM * NHID * 2);
  __bf16* Wt2r = (__bf16*)walloc((size_t)HDIM * NHID * 2);
  float* xl    = (float*)walloc((size_t)NNODE * HDIM * 4);
  float* xr    = (float*)walloc((size_t)NNODE * HDIM * 4);
  float* accum = (float*)walloc((size_t)NNODE * HDIM * 4);
  float* lbuf  = (float*)walloc((size_t)ETOT * NHEADS * 4);
  float* maxb  = (float*)walloc((size_t)NNODE * NHEADS * 4);
  float* den   = (float*)walloc((size_t)NNODE * NHEADS * 4);
  float* h1    = (float*)walloc((size_t)NNODE * NHID * 4);
  float* h2    = (float*)walloc((size_t)NNODE * NHID * 4);
  float* pooled= (float*)walloc((size_t)NBATCH * NHID * 4);
  float* cnt   = (float*)walloc((size_t)NBATCH * 4);

  // ---- prep: cast features, transpose+cast weights ----
  k_cast_bf16<<<cdiv(NNODE * NFEAT, 256), 256, 0, stream>>>(x, xbf, NNODE * NFEAT);
  k_transpose_cast<<<cdiv(NFEAT * HDIM, 256), 256, 0, stream>>>(W1l, Wt1l, NFEAT, HDIM);
  k_transpose_cast<<<cdiv(NFEAT * HDIM, 256), 256, 0, stream>>>(W1r, Wt1r, NFEAT, HDIM);
  k_transpose_cast<<<cdiv(NHID * HDIM, 256), 256, 0, stream>>>(W2l, Wt2l, NHID, HDIM);
  k_transpose_cast<<<cdiv(NHID * HDIM, 256), 256, 0, stream>>>(W2r, Wt2r, NHID, HDIM);

  // ---- layer 1 ----
  run_gat_layer(xbf, NFEAT, Wt1l, b1l, Wt1r, b1r, att1, bias1, ei,
                xl, xr, accum, lbuf, maxb, den, h1, stream);

  // ---- layer 2 ----
  k_cast_bf16<<<cdiv(NNODE * NHID, 256), 256, 0, stream>>>(h1, hbf, NNODE * NHID);
  run_gat_layer(hbf, NHID, Wt2l, b2l, Wt2r, b2r, att2, bias2, ei,
                xl, xr, accum, lbuf, maxb, den, h2, stream);

  // ---- pool + heads ----
  k_fill<<<cdiv(NBATCH * NHID, 256), 256, 0, stream>>>(pooled, 0.0f, NBATCH * NHID);
  k_fill<<<1, 64, 0, stream>>>(cnt, 0.0f, NBATCH);
  k_pool<<<cdiv(NNODE * NHID, 256), 256, 0, stream>>>(h2, batch, pooled, cnt);
  k_head<<<NBATCH, 32, 0, stream>>>(pooled, cnt, Wc, bc, Wv, bv, out);
}
